// ImplicitGenerator3d_6296422056406
// MI455X (gfx1250) — matrix operand order repaired and checked by reading, not compiled, not run on hardware
//
#include <hip/hip_runtime.h>
#include <hip/hip_bf16.h>
#include <math.h>

// ---------------------------------------------------------------------------
// pi-GAN ImplicitGenerator3d forward for gfx1250 (MI455X).
// Heavy SIREN layers run on v_wmma_f32_16x16x32_f16 with LDS-resident
// activations; weights are pre-packed into B-fragment order in workspace.
// ---------------------------------------------------------------------------

typedef __attribute__((ext_vector_type(16))) _Float16 v16h;
typedef __attribute__((ext_vector_type(8)))  _Float16 v8h;
typedef __attribute__((ext_vector_type(8)))  float    v8f;

#define B_N     4
#define W_IMG   32
#define S_N     12
#define NRAY    1024            // W_IMG*W_IMG
#define M_ROWS  (NRAY * S_N)    // 12288 rows per batch
#define HID     256
#define NFILM   8
#define MAPOUT  4608            // (NFILM+1)*HID*2
#define MTILE   32              // rows of activations per block
#define PADROW  264             // 256 + 8 halves pad (keeps 16B alignment)
#define LAYER_HALVES 65536      // 8ks * 16ct * 32lane * 16halves

// ---------------------------------------------------------------------------
// Pack w_hid (7 layers) + w_cs[3:,:] (layer 7) into f16 B-fragment layout:
// packW[layer][ks][ct][lane][j]; n = ct*16 + (lane&15); k = ks*32 + (lane>>4)*16 + j
// ---------------------------------------------------------------------------
__global__ void packw_kernel(const float* __restrict__ w_hid,
                             const float* __restrict__ w_cs,
                             _Float16* __restrict__ packW) {
    int id = blockIdx.x * blockDim.x + threadIdx.x;
    if (id >= 8 * LAYER_HALVES) return;
    int j     = id & 15;
    int lane  = (id >> 4) & 31;
    int ct    = (id >> 9) & 15;
    int ks    = (id >> 13) & 7;
    int layer = id >> 16;
    int n = ct * 16 + (lane & 15);
    int k = ks * 32 + (lane >> 4) * 16 + j;
    float w = (layer < 7) ? w_hid[((size_t)layer * 256 + k) * 256 + n]
                          : w_cs[(size_t)(3 + k) * 256 + n];
    packW[id] = (_Float16)w;
}

// ---------------------------------------------------------------------------
// Mapping network layer: out[b,n] = act(in[b,:] @ W[:,n] + bias[n])
// ---------------------------------------------------------------------------
__global__ void map_layer_kernel(const float* __restrict__ in,
                                 const float* __restrict__ W,
                                 const float* __restrict__ bias,
                                 float* __restrict__ out,
                                 int K, int N, int act) {
    int id = blockIdx.x * blockDim.x + threadIdx.x;
    if (id >= B_N * N) return;
    int b = id / N, n = id - b * N;
    float s = bias[n];
    for (int k = 0; k < K; ++k) s += in[b * K + k] * W[(size_t)k * N + n];
    if (act) s = (s > 0.f) ? s : 0.2f * s;
    out[id] = s;
}

// ---------------------------------------------------------------------------
// Camera + ray setup. t_points = origin + t_dirs * z  (since R(d*z)=Rd*z).
// Writes t_dirs, origins, perturbed coarse z, and pitch/yaw into d_out tail.
// ---------------------------------------------------------------------------
__global__ void cam_kernel(const float* __restrict__ cam_noise,
                           const float* __restrict__ perturb_u,
                           float* __restrict__ tdirs,
                           float* __restrict__ origins,
                           float* __restrict__ zbuf,
                           float* __restrict__ out_tail) {
    int id = blockIdx.x * blockDim.x + threadIdx.x;
    if (id >= B_N * NRAY) return;
    int b = id >> 10, ray = id & 1023;

    const float PI   = 3.14159265358979f;
    const float HPI  = 1.57079632679490f;
    float theta = cam_noise[b * 2 + 0] * 0.3f + HPI;
    float phi   = cam_noise[b * 2 + 1] * 0.155f + HPI;
    phi = fminf(fmaxf(phi, 1e-5f), PI - 1e-5f);

    float sp = __sinf(phi), cp = __cosf(phi);
    float st = __sinf(theta), cth = __cosf(theta);
    float ox = sp * cth, oy = cp, oz = sp * st;          // |origin| == 1
    float fx = -ox, fy = -oy, fz = -oz;                  // fwd = normalize(-origin)
    // left = normalize(cross(up, fwd)), up=(0,1,0)
    float lx = fz, lz = -fx;
    float inv = 1.f / (sqrtf(lx * lx + lz * lz) + 1e-12f);
    lx *= inv; lz *= inv;
    // up2 = normalize(cross(fwd, left)) (left.y == 0)
    float ux = fy * lz, uy = fz * lx - fx * lz, uz = -fy * lx;
    float inv2 = 1.f / (sqrtf(ux * ux + uy * uy + uz * uz) + 1e-12f);
    ux *= inv2; uy *= inv2; uz *= inv2;

    // camera-space ray dir
    int iy = ray >> 5, ix = ray & 31;
    float xs = -1.f + 2.f * (float)ix / 31.f;
    float ys = -1.f + 2.f * (float)iy / 31.f;
    float zc = -1.f / tanf(PI * 12.f / 360.f);
    float invn = 1.f / (sqrtf(xs * xs + ys * ys + zc * zc) + 1e-12f);
    float dcx = xs * invn, dcy = ys * invn, dcz = zc * invn;

    // t_dir = R @ d_cam, columns of R = (-left, up2, -fwd)
    tdirs[id * 3 + 0] = -lx * dcx + ux * dcy - fx * dcz;
    tdirs[id * 3 + 1] =  0.f      + uy * dcy - fy * dcz;
    tdirs[id * 3 + 2] = -lz * dcx + uz * dcy - fz * dcz;

    const float z0 = 0.88f;
    const float dz = (1.12f - 0.88f) / 11.f;
    for (int s = 0; s < S_N; ++s) {
        float u = perturb_u[(size_t)id * S_N + s];
        zbuf[(size_t)id * S_N + s] = z0 + dz * (float)s + (u - 0.5f) * dz;
    }
    if (ray == 0) {
        origins[b * 3 + 0] = ox; origins[b * 3 + 1] = oy; origins[b * 3 + 2] = oz;
        out_tail[b * 2 + 0] = phi;    // pitch
        out_tail[b * 2 + 1] = theta;  // yaw
    }
}

// ---------------------------------------------------------------------------
// Dray[b,ray,c] = dot(t_dirs[b,ray,:], w_cs[0:3,c])   (dirs part of cs layer)
// ---------------------------------------------------------------------------
__global__ void dray_kernel(const float* __restrict__ tdirs,
                            const float* __restrict__ w_cs,
                            float* __restrict__ Dray) {
    int id = blockIdx.x * blockDim.x + threadIdx.x;
    if (id >= B_N * NRAY * HID) return;
    int c = id & 255, rg = id >> 8;
    float dx = tdirs[rg * 3 + 0], dy = tdirs[rg * 3 + 1], dzv = tdirs[rg * 3 + 2];
    Dray[id] = dx * w_cs[c] + dy * w_cs[256 + c] + dzv * w_cs[512 + c];
}

// ---------------------------------------------------------------------------
// Fused SIREN evaluation with WMMA. One block = MTILE rows x 256 cols.
// Activations double-buffered in LDS; weights streamed from packed global.
// ---------------------------------------------------------------------------
__global__ __launch_bounds__(256) void siren_wmma_kernel(
    const float* __restrict__ zbuf,       // B*NRAY*S  sample depths
    const float* __restrict__ tdirs,      // B*NRAY*3
    const float* __restrict__ origins,    // B*3
    const float* __restrict__ mapout,     // B*MAPOUT  (freqs | phases)
    const _Float16* __restrict__ packW,   // 8 layers of packed f16 weights
    const float* __restrict__ w_in, const float* __restrict__ b_in,
    const float* __restrict__ b_hid, const float* __restrict__ b_cs,
    const float* __restrict__ Dray,
    const float* __restrict__ w_sigma, const float* __restrict__ b_sigma,
    const float* __restrict__ w_c, const float* __restrict__ b_c,
    float* __restrict__ out)              // B*M_ROWS*4 rgb_sigma
{
    __shared__ _Float16 sx[2][MTILE][PADROW];

    const int b    = blockIdx.x / (M_ROWS / MTILE);
    const int tile = blockIdx.x % (M_ROWS / MTILE);
    const int tid  = threadIdx.x;
    const float* freqs  = mapout + (size_t)b * MAPOUT;
    const float* phases = freqs + MAPOUT / 2;
    const float ox = origins[b * 3 + 0];
    const float oy = origins[b * 3 + 1];
    const float oz = origins[b * 3 + 2];

    // ---- input layer (K=3, VALU) with FiLM slice 0 ----
    for (int e = tid; e < MTILE * HID; e += 256) {
        const int rl = e >> 8, c = e & 255;
        const int r  = tile * MTILE + rl;
        const int ray = r / S_N, stp = r - ray * S_N;
        const float zz = zbuf[((size_t)b * NRAY + ray) * S_N + stp];
        const float dx = tdirs[((size_t)b * NRAY + ray) * 3 + 0];
        const float dy = tdirs[((size_t)b * NRAY + ray) * 3 + 1];
        const float dzv = tdirs[((size_t)b * NRAY + ray) * 3 + 2];
        const float px = ox + dx * zz, py = oy + dy * zz, pz = oz + dzv * zz;
        float pre = px * w_in[c] + py * w_in[256 + c] + pz * w_in[512 + c] + b_in[c];
        const float f = freqs[c] * 15.f + 30.f;
        sx[0][rl][c] = (_Float16)__sinf(f * pre + phases[c]);
    }

    const int wave  = tid >> 5, lane = tid & 31;
    const int strip = wave >> 2;        // 0..1: 16-row strip
    const int ct0   = (wave & 3) * 4;   // 4 col-tiles per wave
    const int m_a   = strip * 16 + (lane & 15);
    const int kh    = lane >> 4;
    const int n_l   = lane & 15;
    const int mv0   = strip * 16 + 8 * kh;

    // ---- 7 hidden FiLM layers + cs layer (all 256x256 GEMMs via WMMA) ----
    for (int layer = 0; layer < 8; ++layer) {
        __syncthreads();
        const int cur = layer & 1, nxt = cur ^ 1;

        v8f acc[4];
        #pragma unroll
        for (int t = 0; t < 4; ++t)
            #pragma unroll
            for (int j = 0; j < 8; ++j) acc[t][j] = 0.0f;

        const _Float16* WL = packW + (size_t)layer * LAYER_HALVES;

        #pragma unroll
        for (int ks = 0; ks < 8; ++ks) {
            // A fragment (16x32 f16): contiguous k = kh*16 + j (as packW), two 16B LDS loads
            v8h alo = *(const v8h*)&sx[cur][m_a][ks * 32 + kh * 16];
            v8h ahi = *(const v8h*)&sx[cur][m_a][ks * 32 + kh * 16 + 8];
            v16h a;
            #pragma unroll
            for (int j = 0; j < 8; ++j) { a[j] = alo[j]; a[j + 8] = ahi[j]; }

            #pragma unroll
            for (int t = 0; t < 4; ++t) {
                const _Float16* bp = WL + (((ks * 16 + (ct0 + t)) * 32 + lane) * 16);
                v8h blo = *(const v8h*)bp;
                v8h bhi = *(const v8h*)(bp + 8);
                v16h bb;
                #pragma unroll
                for (int j = 0; j < 8; ++j) { bb[j] = blo[j]; bb[j + 8] = bhi[j]; }
                acc[t] = __builtin_amdgcn_wmma_f32_16x16x32_f16(
                    false, a, false, bb, (short)0, acc[t], false, false);
            }
        }

        // ---- FiLM epilogue: x = sin(f*(h+bias[+Dray]) + phase) -> LDS ----
        const int s_off = (layer + 1) * 256;   // FiLM slices 1..8
        #pragma unroll
        for (int t = 0; t < 4; ++t) {
            const int c = (ct0 + t) * 16 + n_l;
            const float f  = freqs[s_off + c] * 15.f + 30.f;
            const float ph = phases[s_off + c];
            const float bias = (layer < 7) ? b_hid[layer * 256 + c] : b_cs[c];
            #pragma unroll
            for (int v = 0; v < 8; ++v) {
                const int ml = mv0 + v;
                float h = acc[t][v] + bias;
                if (layer == 7) {
                    const int r = tile * MTILE + ml;
                    const int ray = r / S_N;
                    h += Dray[((size_t)b * NRAY + ray) * HID + c];
                }
                sx[nxt][ml][c] = (_Float16)__sinf(f * h + ph);
            }
        }
    }
    __syncthreads();

    // ---- heads: sigma (256->1) from x_final (buf 1), rgb (256->3) from
    // rgb_pre (buf 0), sigmoid, write rgb_sigma ----
    if (tid < MTILE) {
        const int rl = tid, r = tile * MTILE + rl;
        float sg = b_sigma[0];
        float c0 = b_c[0], c1 = b_c[1], c2 = b_c[2];
        for (int k = 0; k < 256; ++k) {
            float xv = (float)sx[1][rl][k];
            float rv = (float)sx[0][rl][k];
            sg += xv * w_sigma[k];
            c0 += rv * w_c[k * 3 + 0];
            c1 += rv * w_c[k * 3 + 1];
            c2 += rv * w_c[k * 3 + 2];
        }
        float* o = out + ((size_t)b * M_ROWS + r) * 4;
        o[0] = 1.f / (1.f + __expf(-c0));
        o[1] = 1.f / (1.f + __expf(-c1));
        o[2] = 1.f / (1.f + __expf(-c2));
        o[3] = sg;
    }
}

// ---------------------------------------------------------------------------
// Coarse integration: raw weights[s] = alpha[s] * T
// ---------------------------------------------------------------------------
__global__ void cweights_kernel(const float* __restrict__ rgbsig,
                                const float* __restrict__ zbuf,
                                float* __restrict__ wout) {
    int id = blockIdx.x * blockDim.x + threadIdx.x;
    if (id >= B_N * NRAY) return;
    int b = id >> 10, ray = id & 1023;
    float z[S_N];
    for (int s = 0; s < S_N; ++s) z[s] = zbuf[(size_t)id * S_N + s];
    float T = 1.f;
    for (int s = 0; s < S_N; ++s) {
        float sig = rgbsig[(((size_t)b * M_ROWS) + ray * S_N + s) * 4 + 3];
        float delta = (s < S_N - 1) ? (z[s + 1] - z[s]) : 1e10f;
        float a = 1.f - __expf(-delta * fmaxf(sig, 0.f));
        wout[(size_t)id * S_N + s] = a * T;
        T *= (1.f - a + 1e-10f);
    }
}

// ---------------------------------------------------------------------------
// Hierarchical inverse-CDF sampling (searchsorted right over cdf of 10 bins)
// ---------------------------------------------------------------------------
__global__ void pdf_kernel(const float* __restrict__ wraw,
                           const float* __restrict__ zbuf,
                           const float* __restrict__ pdf_u,
                           float* __restrict__ finez) {
    int id = blockIdx.x * blockDim.x + threadIdx.x;
    if (id >= B_N * NRAY) return;
    float zv[S_N];
    for (int s = 0; s < S_N; ++s) zv[s] = zbuf[(size_t)id * S_N + s];
    float zm[11];
    for (int i = 0; i < 11; ++i) zm[i] = 0.5f * (zv[i] + zv[i + 1]);
    float w[10], sum = 0.f;
    for (int i = 0; i < 10; ++i) {  // weights + 1e-5 (outer) + 1e-5 (sample_pdf)
        w[i] = wraw[(size_t)id * S_N + 1 + i] + 2e-5f;
        sum += w[i];
    }
    float cdf[11]; cdf[0] = 0.f;
    float invs = 1.f / sum;
    for (int i = 0; i < 10; ++i) cdf[i + 1] = cdf[i] + w[i] * invs;
    for (int s = 0; s < S_N; ++s) {
        float u = pdf_u[(size_t)id * S_N + s];
        int ind = 0;
        for (int i = 0; i < 11; ++i) ind += (cdf[i] <= u) ? 1 : 0;
        int below = min(max(ind - 1, 0), 10);
        int above = min(ind, 10);
        float c0 = cdf[below], c1 = cdf[above];
        float b0 = zm[below],  b1 = zm[above];
        float den = c1 - c0;
        if (den < 1e-5f) den = 1.f;
        finez[(size_t)id * S_N + s] = b0 + (u - c0) / den * (b1 - b0);
    }
}

// ---------------------------------------------------------------------------
// Merge fine+coarse (stable sort by z), composite, write pixels (B,3,W,W)
// ---------------------------------------------------------------------------
__global__ void final_kernel(const float* __restrict__ rs_fine,
                             const float* __restrict__ rs_coarse,
                             const float* __restrict__ finez,
                             const float* __restrict__ zc,
                             float* __restrict__ out) {
    int id = blockIdx.x * blockDim.x + threadIdx.x;
    if (id >= B_N * NRAY) return;
    int b = id >> 10, ray = id & 1023;

    float z[24]; int idx[24];
    for (int s = 0; s < S_N; ++s) { z[s] = finez[(size_t)id * S_N + s]; idx[s] = s; }
    for (int s = 0; s < S_N; ++s) { z[12 + s] = zc[(size_t)id * S_N + s]; idx[12 + s] = 12 + s; }
    // stable insertion sort ascending (matches stable argsort: fine before coarse)
    for (int i = 1; i < 24; ++i) {
        float kz = z[i]; int ki = idx[i]; int j = i - 1;
        while (j >= 0 && z[j] > kz) { z[j + 1] = z[j]; idx[j + 1] = idx[j]; --j; }
        z[j + 1] = kz; idx[j + 1] = ki;
    }
    float T = 1.f, r = 0.f, g = 0.f, bl = 0.f;
    for (int s = 0; s < 24; ++s) {
        int k = idx[s];
        const float* rsp = (k < 12)
            ? (rs_fine   + (((size_t)b * M_ROWS) + ray * S_N + k) * 4)
            : (rs_coarse + (((size_t)b * M_ROWS) + ray * S_N + (k - 12)) * 4);
        float delta = (s < 23) ? (z[s + 1] - z[s]) : 1e10f;
        float a = 1.f - __expf(-delta * fmaxf(rsp[3], 0.f));
        float wgt = a * T;
        T *= (1.f - a + 1e-10f);
        r += wgt * rsp[0]; g += wgt * rsp[1]; bl += wgt * rsp[2];
    }
    out[((size_t)b * 3 + 0) * NRAY + ray] = r * 2.f - 1.f;
    out[((size_t)b * 3 + 1) * NRAY + ray] = g * 2.f - 1.f;
    out[((size_t)b * 3 + 2) * NRAY + ray] = bl * 2.f - 1.f;
}

// ---------------------------------------------------------------------------
extern "C" void kernel_launch(void* const* d_in, const int* in_sizes, int n_in,
                              void* d_out, int out_size, void* d_ws, size_t ws_size,
                              hipStream_t stream) {
    const float* zlat      = (const float*)d_in[0];
    const float* perturb_u = (const float*)d_in[1];
    const float* cam_noise = (const float*)d_in[2];
    const float* pdf_u     = (const float*)d_in[3];
    const float* mw0 = (const float*)d_in[4];  const float* mb0 = (const float*)d_in[5];
    const float* mw1 = (const float*)d_in[6];  const float* mb1 = (const float*)d_in[7];
    const float* mw2 = (const float*)d_in[8];  const float* mb2 = (const float*)d_in[9];
    const float* mw_out = (const float*)d_in[10]; const float* mb_out = (const float*)d_in[11];
    const float* w_in  = (const float*)d_in[12]; const float* b_in  = (const float*)d_in[13];
    const float* w_hid = (const float*)d_in[14]; const float* b_hid = (const float*)d_in[15];
    const float* w_sig = (const float*)d_in[16]; const float* b_sig = (const float*)d_in[17];
    const float* w_cs  = (const float*)d_in[18]; const float* b_cs  = (const float*)d_in[19];
    const float* w_c   = (const float*)d_in[20]; const float* b_c   = (const float*)d_in[21];
    float* out = (float*)d_out;

    // -------- workspace carving (256B aligned) --------
    char* wsp = (char*)d_ws;
    size_t off = 0;
    auto carve = [&](size_t bytes) -> void* {
        void* p = wsp + off;
        off += (bytes + 255) & ~(size_t)255;
        return p;
    };
    float* map_h0  = (float*)carve((size_t)B_N * HID * 4);
    float* map_h1  = (float*)carve((size_t)B_N * HID * 4);
    float* map_out = (float*)carve((size_t)B_N * MAPOUT * 4);
    float* tdirs   = (float*)carve((size_t)B_N * NRAY * 3 * 4);
    float* origins = (float*)carve((size_t)B_N * 3 * 4);
    float* z_crs   = (float*)carve((size_t)B_N * NRAY * S_N * 4);
    float* z_fine  = (float*)carve((size_t)B_N * NRAY * S_N * 4);
    float* Dray    = (float*)carve((size_t)B_N * NRAY * HID * 4);
    _Float16* packW = (_Float16*)carve((size_t)8 * LAYER_HALVES * 2);
    float* rs_c    = (float*)carve((size_t)B_N * M_ROWS * 4 * 4);
    float* rs_f    = (float*)carve((size_t)B_N * M_ROWS * 4 * 4);
    float* wraw    = (float*)carve((size_t)B_N * NRAY * S_N * 4);
    (void)ws_size; (void)in_sizes; (void)n_in; (void)out_size;

    // -------- weight prepack + mapping network --------
    packw_kernel<<<(8 * LAYER_HALVES + 255) / 256, 256, 0, stream>>>(w_hid, w_cs, packW);
    map_layer_kernel<<<(B_N * HID + 255) / 256, 256, 0, stream>>>(zlat, mw0, mb0, map_h0, 256, 256, 1);
    map_layer_kernel<<<(B_N * HID + 255) / 256, 256, 0, stream>>>(map_h0, mw1, mb1, map_h1, 256, 256, 1);
    map_layer_kernel<<<(B_N * HID + 255) / 256, 256, 0, stream>>>(map_h1, mw2, mb2, map_h0, 256, 256, 1);
    map_layer_kernel<<<(B_N * MAPOUT + 255) / 256, 256, 0, stream>>>(map_h0, mw_out, mb_out, map_out, 256, MAPOUT, 0);

    // -------- rays / camera / dirs-part of cs layer --------
    cam_kernel<<<(B_N * NRAY + 255) / 256, 256, 0, stream>>>(
        cam_noise, perturb_u, tdirs, origins, z_crs, out + (size_t)B_N * 3 * NRAY);
    dray_kernel<<<(B_N * NRAY * HID + 255) / 256, 256, 0, stream>>>(tdirs, w_cs, Dray);

    const int siren_blocks = B_N * (M_ROWS / MTILE);  // 4 * 384 = 1536

    // -------- coarse SIREN pass (WMMA) --------
    siren_wmma_kernel<<<siren_blocks, 256, 0, stream>>>(
        z_crs, tdirs, origins, map_out, packW, w_in, b_in, b_hid, b_cs, Dray,
        w_sig, b_sig, w_c, b_c, rs_c);

    // -------- hierarchical sampling --------
    cweights_kernel<<<(B_N * NRAY + 255) / 256, 256, 0, stream>>>(rs_c, z_crs, wraw);
    pdf_kernel<<<(B_N * NRAY + 255) / 256, 256, 0, stream>>>(wraw, z_crs, pdf_u, z_fine);

    // -------- fine SIREN pass (WMMA) --------
    siren_wmma_kernel<<<siren_blocks, 256, 0, stream>>>(
        z_fine, tdirs, origins, map_out, packW, w_in, b_in, b_hid, b_cs, Dray,
        w_sig, b_sig, w_c, b_c, rs_f);

    // -------- merge + composite + output --------
    final_kernel<<<(B_N * NRAY + 255) / 256, 256, 0, stream>>>(rs_f, rs_c, z_fine, z_crs, out);
}
